// RelativeMultiHeadAttn_10788957847947
// MI455X (gfx1250) — matrix operand order repaired and checked by reading, not compiled, not run on hardware
//
#include <hip/hip_runtime.h>

// ---------------------------------------------------------------------------
// Relative multi-head attention (Transformer-XL style), MI455X / gfx1250.
// B=8, S=512, D_MODEL=1024, N_HEAD=16, HEAD_DIM=64.
//
// logit[b,h,q,k] = (q + r_r_bias[h]) . K[b,h,k]
//               + (q + r_w_bias[h]) . pos[S + k - q]        (shift identity)
// out = softmax(logit) @ V
//
// All matrix math in bf16 WMMA (v_wmma_f32_16x16x32_bf16), f32 accumulate.
// GEMM main loop uses CDNA5 async global->LDS copies (ASYNCcnt) when the
// toolchain exposes the builtins; guarded fallback otherwise.
// ---------------------------------------------------------------------------

#define Sq 512
#define HD 64
#define NH 16
#define NB 8
#define DM 1024
#define POS_COLS 1104   // 1 left pad + 1024 valid + right pad (zeros)

typedef __attribute__((ext_vector_type(16))) __bf16   v16bf;
typedef __attribute__((ext_vector_type(8)))  float    v8f;
typedef __attribute__((ext_vector_type(4)))  unsigned uu4;
typedef __attribute__((ext_vector_type(4)))  int      v4i;

union U16bf { uu4 q[2]; v16bf v; unsigned u[8]; };

#if defined(__gfx1250__) && __has_builtin(__builtin_amdgcn_global_load_async_to_lds_b128)
#define HAVE_ASYNC_LDS 1
#else
#define HAVE_ASYNC_LDS 0
#endif

// pointer-to-AS1/AS3 int4, matching the async builtin's parameter types
typedef __attribute__((address_space(1))) v4i* gp4_t;
typedef __attribute__((address_space(3))) v4i* lp4_t;

__device__ __forceinline__ v8f wmma_bf16(v16bf a, v16bf b, v8f c) {
  // 8 args: (neg_a, A, neg_b, B, c_mod, C, reuse_a, reuse_b)
  return __builtin_amdgcn_wmma_f32_16x16x32_bf16(false, a, false, b, (short)0, c,
                                                 false, false);
}

template <int N>
__device__ __forceinline__ void wait_async() {
#if __has_builtin(__builtin_amdgcn_s_wait_asynccnt)
  __builtin_amdgcn_s_wait_asynccnt(N);
#else
  asm volatile("s_wait_asynccnt %0" ::"i"(N) : "memory");
#endif
}

// async 16B global->LDS copy (per lane); falls back to a synchronous copy.
__device__ __forceinline__ void cp16_g2l(const __bf16* g, __bf16* l) {
#if HAVE_ASYNC_LDS
  __builtin_amdgcn_global_load_async_to_lds_b128(
      (gp4_t)(unsigned long long)g,
      (lp4_t)(unsigned)(unsigned long long)l,
      0, 0);
#else
  *(uu4*)l = *(const uu4*)g;
#endif
}

// Load a 16x32 bf16 A-fragment from a row-major matrix (ld elems/row).
// CDNA5 A layout: lane L -> row m=L&15; lanes<16 hold K={0..7,16..23},
// lanes>=16 hold K={8..15,24..31}, 2 elems per VGPR.  16B-aligned reads.
__device__ __forceinline__ v16bf ldA(const __bf16* src, int ld, int r0, int c0,
                                     int lane) {
  int m  = lane & 15;
  int kb = (lane & 16) ? 8 : 0;
  const __bf16* p = src + (long)(r0 + m) * ld + c0 + kb;
  U16bf t;
  t.q[0] = *(const uu4*)(p);
  t.q[1] = *(const uu4*)(p + 16);
  return t.v;
}

// Load a 32x16 bf16 B-fragment: lane L -> col n=L&15; lanes<16 hold
// K={0..7,16..23}, lanes>=16 hold K={8..15,24..31}.
__device__ __forceinline__ v16bf ldB(const __bf16* src, int ld, int r0, int c0,
                                     int lane) {
  const __bf16* p = src + (long)(r0 + ((lane & 16) ? 8 : 0)) * ld + c0 + (lane & 15);
  v16bf t;
#pragma unroll
  for (int s = 0; s < 16; ++s) t[s] = p[(long)(s < 8 ? s : s + 8) * ld];
  return t;
}

// Same layout for the pos table band reads (row stride POS_COLS).
__device__ __forceinline__ v16bf ldBu(const __bf16* p) {
  v16bf t;
#pragma unroll
  for (int s = 0; s < 16; ++s) t[s] = p[(long)(s < 8 ? s : s + 8) * POS_COLS];
  return t;
}

// ---------------------------------------------------------------------------
// Kernel 1: transposed sinusoidal position table, bf16, padded.
// posT[d][1+l] = (d<32 ? sin : cos)((l-512) * exp(-(d&31)*ln(1e4)/31))
// ---------------------------------------------------------------------------
__global__ void pos_kernel(__bf16* posT) {
  int i = blockIdx.x * 256 + threadIdx.x;
  if (i >= 64 * POS_COLS) return;
  int c = i % POS_COLS;
  int d = i / POS_COLS;
  float v = 0.f;
  int l = c - 1;
  if (l >= 0 && l < 2 * Sq) {
    float freq = __expf(-0.29710775393471563f * (float)(d & 31));
    float ang  = (float)(l - Sq) * freq;
    v = (d < 32) ? sinf(ang) : cosf(ang);
  }
  posT[i] = (__bf16)v;
}

// ---------------------------------------------------------------------------
// Kernel 2a: x staging.  xbf = bf16(x) row-major (GEMM A operand),
//            kt[((b*16+h)*64+d)*512+s] = bf16(x[b,s,h*64+d])  (K^T per head)
// ---------------------------------------------------------------------------
__global__ void stage_x_kernel(const float* __restrict__ x,
                               __bf16* __restrict__ xbf,
                               __bf16* __restrict__ kt) {
  long i = (long)blockIdx.x * 256 + threadIdx.x;  // 2^22 elements
  float v = x[i];
  xbf[i] = (__bf16)v;
  int c = (int)(i & 1023);
  int s = (int)((i >> 10) & 511);
  int b = (int)(i >> 19);
  int h = c >> 6, d = c & 63;
  kt[((long)(b * NH + h) * HD + d) * Sq + s] = (__bf16)v;
}

// Kernel 2b: Wqv -> bf16
__global__ void stage_w_kernel(const float* __restrict__ w,
                               __bf16* __restrict__ wbf) {
  long i = (long)blockIdx.x * 256 + threadIdx.x;  // 2^21 elements
  wbf[i] = (__bf16)w[i];
}

// ---------------------------------------------------------------------------
// Kernel 3: qv = x @ Wqv GEMM (M=4096,N=2048,K=1024), bf16 WMMA.
// Double-buffered LDS tiles fed by async global->LDS (ASYNCcnt pipeline).
// Epilogue (block-uniform q/v routing) scatters into head-major bf16 buffers
// with fused bias adds: n<1024: qr=q+r_r_bias, qw=q+r_w_bias; else vb=v.
// ---------------------------------------------------------------------------
__device__ __forceinline__ void issue_tile(const __bf16* __restrict__ xbf,
                                           const __bf16* __restrict__ wbf,
                                           int m0, int n0, int kk,
                                           __bf16 (*A)[40], __bf16 (*B)[72],
                                           int tid) {
  // A tile 128x32: 512 x 16B chunks, 2 per thread
#pragma unroll
  for (int t = 0; t < 2; ++t) {
    int c = tid + t * 256;
    int row = c >> 2, col = (c & 3) * 8;
    cp16_g2l(xbf + (long)(m0 + row) * DM + kk + col, &A[row][col]);
  }
  // B tile 32x64: 256 x 16B chunks, 1 per thread
  {
    int row = tid >> 3, col = (tid & 7) * 8;
    cp16_g2l(wbf + (long)(kk + row) * 2048 + n0 + col, &B[row][col]);
  }
}

__global__ __launch_bounds__(256) void qv_gemm_kernel(
    const __bf16* __restrict__ xbf, const __bf16* __restrict__ wbf,
    const float* __restrict__ rrb, const float* __restrict__ rwb,
    __bf16* __restrict__ qr, __bf16* __restrict__ qw, __bf16* __restrict__ vb) {
  __shared__ __align__(16) __bf16 A_lds[2][128][40];  // padded rows (80B = 5x16)
  __shared__ __align__(16) __bf16 B_lds[2][32][72];   // padded rows (144B = 9x16)

  int tid  = threadIdx.x;
  int wave = tid >> 5;
  int lane = tid & 31;
  int bm = blockIdx.x >> 5;          // 32 M-blocks of 128
  int bn = blockIdx.x & 31;          // 32 N-blocks of 64
  int m0 = bm * 128, n0 = bn * 64;
  int wm = (wave >> 1) * 32;         // 4 waves along M
  int wn = (wave & 1) * 32;          // 2 waves along N

  v8f acc[2][2] = {};

  issue_tile(xbf, wbf, m0, n0, 0, A_lds[0], B_lds[0], tid);

  for (int i = 0; i < 32; ++i) {
    int cur = i & 1;
    if (i < 31) {
      issue_tile(xbf, wbf, m0, n0, (i + 1) * 32, A_lds[cur ^ 1], B_lds[cur ^ 1],
                 tid);
      wait_async<3>();   // per-wave: 3 newer async ops pending => tile i landed
    } else {
      wait_async<0>();
    }
    __syncthreads();

    v16bf a0 = ldA(&A_lds[cur][0][0], 40, wm + 0,  0, lane);
    v16bf a1 = ldA(&A_lds[cur][0][0], 40, wm + 16, 0, lane);
    v16bf b0 = ldB(&B_lds[cur][0][0], 72, 0, wn + 0,  lane);
    v16bf b1 = ldB(&B_lds[cur][0][0], 72, 0, wn + 16, lane);
    acc[0][0] = wmma_bf16(a0, b0, acc[0][0]);
    acc[0][1] = wmma_bf16(a0, b1, acc[0][1]);
    acc[1][0] = wmma_bf16(a1, b0, acc[1][0]);
    acc[1][1] = wmma_bf16(a1, b1, acc[1][1]);
    __syncthreads();   // all waves done reading buf `cur` before it is refilled
  }

  // C layout: lane L, VGPR r -> (m = r + 8*(L>>4), n = L&15).
  // q-vs-v routing is uniform per block (n0 multiple of 64).
  int n = lane & 15, hi = lane >> 4;
  if (n0 < DM) {  // q half: col == h*64+d
#pragma unroll
    for (int mi = 0; mi < 2; ++mi)
#pragma unroll
      for (int ni = 0; ni < 2; ++ni)
#pragma unroll
        for (int r = 0; r < 8; ++r) {
          int row = m0 + wm + mi * 16 + r + 8 * hi;  // b*512+s
          int col = n0 + wn + ni * 16 + n;           // [0,1024)
          int b = row >> 9, s = row & 511;
          int h = col >> 6;
          long off = (((long)(b * NH + h)) * Sq + s) * HD + (col & 63);
          float a = acc[mi][ni][r];
          qr[off] = (__bf16)(a + rrb[col]);
          qw[off] = (__bf16)(a + rwb[col]);
        }
  } else {  // v half
#pragma unroll
    for (int mi = 0; mi < 2; ++mi)
#pragma unroll
      for (int ni = 0; ni < 2; ++ni)
#pragma unroll
        for (int r = 0; r < 8; ++r) {
          int row = m0 + wm + mi * 16 + r + 8 * hi;
          int col = n0 - DM + wn + ni * 16 + n;      // [0,1024)
          int b = row >> 9, s = row & 511;
          int h = col >> 6;
          vb[(((long)(b * NH + h)) * Sq + s) * HD + (col & 63)] =
              (__bf16)acc[mi][ni][r];
        }
  }
}

// ---------------------------------------------------------------------------
// Kernel 4: fused flash attention with relative-shift band.
// Block = 4 waves; each wave owns 16 q-rows of one (b,h).  Per 32-key step:
//   4 WMMA (Q_r . K^T), 2 WMMA (E band = Q_w . pos^T), softmax, 4 WMMA (P.V)
// E ring: E[m][j mod 64],  score(m, k) += E[m][k + 15 - m]  (j base L0=497-q0)
// ---------------------------------------------------------------------------
__device__ __forceinline__ void computeE(int j0, int lane, const __bf16* pr0,
                                         const __bf16* pr1, v16bf a0, v16bf a1,
                                         float (*E)[64]) {
  v8f e = {};
  e = wmma_bf16(a0, ldBu(pr0 + j0), e);
  e = wmma_bf16(a1, ldBu(pr1 + j0), e);
  int n = lane & 15, mb = (lane >> 4) * 8, col = (j0 & 63) + n;
#pragma unroll
  for (int r = 0; r < 8; ++r) E[mb + r][col] = e[r];
}

__global__ __launch_bounds__(128) void attn_kernel(
    const __bf16* __restrict__ qr, const __bf16* __restrict__ qw,
    const __bf16* __restrict__ kt, const __bf16* __restrict__ vb,
    const __bf16* __restrict__ posT, const int* __restrict__ mask,
    float* __restrict__ out) {
  __shared__ float E_lds[4][16][64];                 // 4-tile ring per wave
  __shared__ __align__(16) __bf16 P_lds[4][16][32];  // P relayout per wave

  int tid = threadIdx.x, wave = tid >> 5, lane = tid & 31;
  int qc = blockIdx.x & 7;        // 8 q-chunks of 64
  int bh = blockIdx.x >> 3;       // b*16 + h
  int b  = bh >> 4;
  int q0 = qc * 64 + wave * 16;

  const __bf16* qr_b = qr + (long)bh * Sq * HD;
  const __bf16* qw_b = qw + (long)bh * Sq * HD;
  const __bf16* kt_b = kt + (long)bh * HD * Sq;
  const __bf16* vb_b = vb + (long)bh * Sq * HD;

  // Q fragments, k-chunks 0..31 / 32..63, kept in registers
  v16bf qrA0 = ldA(qr_b, HD, q0, 0, lane);
  v16bf qrA1 = ldA(qr_b, HD, q0, 32, lane);
  v16bf qwA0 = ldA(qw_b, HD, q0, 0, lane);
  v16bf qwA1 = ldA(qw_b, HD, q0, 32, lane);

  // padded pos column base: l = (497 - q0) + j, +1 left pad -> even, 4B align
  int L0pad = 498 - q0;
  const __bf16* pr0 = posT + (long)((lane & 16) ? 8 : 0) * POS_COLS + L0pad + (lane & 15);        // d base 8h, j = lane&15
  const __bf16* pr1 = posT + (long)(32 + ((lane & 16) ? 8 : 0)) * POS_COLS + L0pad + (lane & 15);   // d base 32+8h
  float(*Ew)[64] = E_lds[wave];

  computeE(0, lane, pr0, pr1, qwA0, qwA1, Ew);
  computeE(16, lane, pr0, pr1, qwA0, qwA1, Ew);

  float mrow[8], drow[8];
  v8f o[4] = {};
#pragma unroll
  for (int r = 0; r < 8; ++r) { mrow[r] = -1e30f; drow[r] = 0.f; }

  int n = lane & 15, hi = lane >> 4;

  for (int k0 = 0; k0 < Sq; k0 += 32) {
    computeE(k0 + 32, lane, pr0, pr1, qwA0, qwA1, Ew);  // extend band ring

    if (k0 + 32 < Sq) {  // hint next K/V tiles toward the caches
      __builtin_prefetch((const void*)(kt_b + (long)lane * Sq + k0 + 32), 0, 0);
      __builtin_prefetch((const void*)(vb_b + (long)(k0 + 32 + lane) * HD), 0, 0);
    }

    // scores: two 16x16 tiles, K-dim 64 via two bf16 WMMAs each
    v8f s0 = {}, s1 = {};
    s0 = wmma_bf16(qrA0, ldB(kt_b, Sq, 0,  k0, lane), s0);
    s0 = wmma_bf16(qrA1, ldB(kt_b, Sq, 32, k0, lane), s0);
    s1 = wmma_bf16(qrA0, ldB(kt_b, Sq, 0,  k0 + 16, lane), s1);
    s1 = wmma_bf16(qrA1, ldB(kt_b, Sq, 32, k0 + 16, lane), s1);

    float mv0 = (float)mask[b * Sq + k0 + n];
    float mv1 = (float)mask[b * Sq + k0 + 16 + n];
#pragma unroll
    for (int r = 0; r < 8; ++r) {
      int m = r + 8 * hi;
      float x0 = s0[r] + Ew[m][(k0 + n + 15 - m) & 63];
      float x1 = s1[r] + Ew[m][(k0 + 16 + n + 15 - m) & 63];
      s0[r] = x0 * mv0 - (1.f - mv0) * 1e8f;
      s1[r] = x1 * mv1 - (1.f - mv1) * 1e8f;
    }

    // online softmax; each row lives on a 16-lane group (8 rows per lane)
#pragma unroll
    for (int r = 0; r < 8; ++r) {
      float mx = fmaxf(s0[r], s1[r]);
      mx = fmaxf(mx, __shfl_xor(mx, 1, 32));
      mx = fmaxf(mx, __shfl_xor(mx, 2, 32));
      mx = fmaxf(mx, __shfl_xor(mx, 4, 32));
      mx = fmaxf(mx, __shfl_xor(mx, 8, 32));
      float mnew  = fmaxf(mrow[r], mx);
      float alpha = __expf(mrow[r] - mnew);
      mrow[r] = mnew;
      float p0 = __expf(s0[r] - mnew);
      float p1 = __expf(s1[r] - mnew);
      float ps = p0 + p1;
      ps += __shfl_xor(ps, 1, 32);
      ps += __shfl_xor(ps, 2, 32);
      ps += __shfl_xor(ps, 4, 32);
      ps += __shfl_xor(ps, 8, 32);
      drow[r] = drow[r] * alpha + ps;
#pragma unroll
      for (int t = 0; t < 4; ++t) o[t][r] *= alpha;
      s0[r] = p0;
      s1[r] = p1;
    }

    // C-tile -> A-fragment relayout of P through LDS (bf16)
#pragma unroll
    for (int r = 0; r < 8; ++r) {
      int m = r + 8 * hi;
      P_lds[wave][m][n]      = (__bf16)s0[r];
      P_lds[wave][m][16 + n] = (__bf16)s1[r];
    }
    v16bf pA;
    {
      int m = lane & 15;
      int kb = (lane & 16) ? 8 : 0;
      U16bf t;
      t.q[0] = *(const uu4*)&P_lds[wave][m][kb];
      t.q[1] = *(const uu4*)&P_lds[wave][m][16 + kb];
      pA = t.v;
    }
#pragma unroll
    for (int t = 0; t < 4; ++t)
      o[t] = wmma_bf16(pA, ldB(vb_b, HD, k0, t * 16, lane), o[t]);
  }

  // epilogue: out[b, q0+m, h*64 + d] = o / denom
  float* outp = out + ((long)(b * Sq + q0)) * DM + (bh & 15) * HD;
#pragma unroll
  for (int t = 0; t < 4; ++t)
#pragma unroll
    for (int r = 0; r < 8; ++r) {
      int m = r + 8 * hi;
      int d = t * 16 + n;
      outp[(long)m * DM + d] = o[t][r] / drow[r];
    }
}

// ---------------------------------------------------------------------------
extern "C" void kernel_launch(void* const* d_in, const int* in_sizes, int n_in,
                              void* d_out, int out_size, void* d_ws,
                              size_t ws_size, hipStream_t stream) {
  (void)in_sizes; (void)n_in; (void)out_size; (void)ws_size;
  const float* x    = (const float*)d_in[0];
  const int*   mask = (const int*)d_in[1];
  const float* Wqv  = (const float*)d_in[2];
  const float* rrb  = (const float*)d_in[3];  // r_r_bias -> AC term
  const float* rwb  = (const float*)d_in[4];  // r_w_bias -> pos term
  float* out = (float*)d_out;

  const size_t HEADS_ELEMS = (size_t)NB * NH * Sq * HD;  // 2^22
  char* ws = (char*)d_ws;
  __bf16* posT = (__bf16*)ws;  ws += ((size_t)64 * POS_COLS * 2 + 255) & ~255ull;
  __bf16* qr  = (__bf16*)ws;   ws += HEADS_ELEMS * 2;
  __bf16* qw  = (__bf16*)ws;   ws += HEADS_ELEMS * 2;
  __bf16* kt  = (__bf16*)ws;   ws += HEADS_ELEMS * 2;
  __bf16* vb  = (__bf16*)ws;   ws += HEADS_ELEMS * 2;
  __bf16* xbf = (__bf16*)ws;   ws += HEADS_ELEMS * 2;                 // 4096x1024
  __bf16* wbf = (__bf16*)ws;   ws += (size_t)DM * 2048 * 2;           // 1024x2048

  pos_kernel<<<(64 * POS_COLS + 255) / 256, 256, 0, stream>>>(posT);
  stage_x_kernel<<<(int)(HEADS_ELEMS / 256), 256, 0, stream>>>(x, xbf, kt);
  stage_w_kernel<<<(int)((size_t)DM * 2048 / 256), 256, 0, stream>>>(Wqv, wbf);
  qv_gemm_kernel<<<32 * 32, 256, 0, stream>>>(xbf, wbf, rrb, rwb, qr, qw, vb);
  attn_kernel<<<NB * NH * (Sq / 64), 128, 0, stream>>>(qr, qw, kt, vb, posT,
                                                       mask, out);
}
